// GeneEncoder_81157702025559
// MI455X (gfx1250) — matrix operand order repaired and checked by reading, not compiled, not run on hardware
//
#include <hip/hip_runtime.h>
#include <hip/hip_bf16.h>
#include <math.h>

typedef __bf16 bf16_t;
typedef __attribute__((ext_vector_type(16))) __bf16 bf16x16;
typedef __attribute__((ext_vector_type(8)))  float  f32x8;

#define NG   20000
#define NC   1024
#define EMBD 512
#define NEDG 640000

// ---------------- elementwise helpers ----------------

__global__ void k_zero(float* p, int n) {
    int i = blockIdx.x * blockDim.x + threadIdx.x;
    if (i < n) p[i] = 0.f;
}

__global__ void k_cvt(const float* __restrict__ in, bf16_t* __restrict__ out, int n) {
    int i = blockIdx.x * blockDim.x + threadIdx.x;
    if (i < n) out[i] = (bf16_t)in[i];
}

__global__ void k_add_cvt(const float* __restrict__ a, const float* __restrict__ b,
                          bf16_t* __restrict__ out, int n) {
    int i = blockIdx.x * blockDim.x + threadIdx.x;
    if (i < n) out[i] = (bf16_t)(a[i] + b[i]);
}

// degree count: one thread per edge
__global__ void k_deg(const int* __restrict__ dst, float* deg, int e) {
    int i = blockIdx.x * blockDim.x + threadIdx.x;
    if (i < e) atomicAdd(&deg[dst[i]], 1.f);
}

// edge-parallel scatter-add: each thread moves 4 consecutive features of one edge.
// F = 1<<fshift; per-edge thread count = 1<<(fshift-2); shift/mask indexing only.
__global__ void k_scatter4(const float* __restrict__ x, const int* __restrict__ src,
                           const int* __restrict__ dst, float* agg, int E, int fshift) {
    const int eshift = fshift - 2;
    long long idx = (long long)blockIdx.x * blockDim.x + threadIdx.x;
    long long total = (long long)E << eshift;
    if (idx >= total) return;
    int e = (int)(idx >> eshift);
    int f = ((int)idx & ((1 << eshift) - 1)) << 2;
    float4 v = *(const float4*)(x + ((size_t)src[e] << fshift) + f);
    float* a = agg + ((size_t)dst[e] << fshift) + f;
    atomicAdd(a + 0, v.x);
    atomicAdd(a + 1, v.y);
    atomicAdd(a + 2, v.z);
    atomicAdd(a + 3, v.w);
}

// mean = agg / max(deg,1), emitted directly as bf16 GEMM operand
__global__ void k_mean_cvt(const float* __restrict__ agg, const float* __restrict__ deg,
                           bf16_t* __restrict__ out, int M, int fshift) {
    int i = blockIdx.x * blockDim.x + threadIdx.x;
    if (i >= (M << fshift)) return;
    int m = i >> fshift;
    float d = deg[m];
    d = d > 1.f ? d : 1.f;
    out[i] = (bf16_t)(agg[i] / d);
}

// ---------------- WMMA GEMM, 64x32 register-blocked ----------------
// C[M,N] = act( A1[M,K1] @ W1[N,K1]^T (+ A2[M,K2] @ W2[N,K2]^T) + bias )
// One wave owns a 64x32 output tile = 4x2 grid of 16x16 WMMA tiles.
// Per 32-K step: 4 A-frags + 2 B-frags feed 8 v_wmma -> 1.5 b128 loads per wmma.
// Per-lane fragment layout used here (A/B contiguous K is this code's choice, not the ISA table order):
//   A: lane (l%16)=row, l/16 selects K 0-15 vs 16-31 (16 contiguous bf16)
//   B: lane (l%16)=col, l/16 selects K 0-15 vs 16-31 (16 contiguous bf16)
//   C/D: vgpr r holds row r + 8*(l/16), col = l%16

union Frag { bf16x16 v; uint4 u[2]; };

__device__ __forceinline__ void mma_pass(const bf16_t* __restrict__ A,
                                         const bf16_t* __restrict__ W, int K,
                                         const int* rows, int n0, int kh,
                                         f32x8 acc[4][2]) {
    const bf16_t* brow0 = W + (size_t)n0 * K;
    const bf16_t* brow1 = W + (size_t)(n0 + 16) * K;
    for (int kb = 0; kb < K; kb += 32) {
        Frag a[4], b[2];
#pragma unroll
        for (int i = 0; i < 4; ++i) {
            const bf16_t* ar = A + (size_t)rows[i] * K + kb + kh * 16;
            a[i].u[0] = *(const uint4*)(ar);
            a[i].u[1] = *(const uint4*)(ar + 8);
        }
        const uint4* bp0 = (const uint4*)(brow0 + kb + kh * 16);
        b[0].u[0] = bp0[0];
        b[0].u[1] = bp0[1];
        const uint4* bp1 = (const uint4*)(brow1 + kb + kh * 16);
        b[1].u[0] = bp1[0];
        b[1].u[1] = bp1[1];
#pragma unroll
        for (int i = 0; i < 4; ++i) {
#pragma unroll
            for (int j = 0; j < 2; ++j) {
                acc[i][j] = __builtin_amdgcn_wmma_f32_16x16x32_bf16(
                    false, a[i].v, false, b[j].v, (short)0, acc[i][j], false, false);
            }
        }
    }
}

__global__ void __launch_bounds__(256) k_wmma_gemm(
    const bf16_t* __restrict__ A1, const bf16_t* __restrict__ W1, int K1,
    const bf16_t* __restrict__ A2, const bf16_t* __restrict__ W2, int K2,
    const float* __restrict__ bias, float* __restrict__ C, bf16_t* __restrict__ Cbf,
    int M, int N, int act) {
    const int lane = threadIdx.x & 31;
    const int wave = threadIdx.x >> 5;
    const int wid  = blockIdx.x * 8 + wave;
    const int ntn  = N >> 5;               // 32-wide n blocks
    const int mtn  = (M + 63) >> 6;        // 64-tall m blocks
    if (wid >= mtn * ntn) return;
    const int nt = wid % ntn;
    const int mt = wid / ntn;
    const int lm = lane & 15;
    const int kh = lane >> 4;

    int rows[4];
#pragma unroll
    for (int i = 0; i < 4; ++i) {
        int r = mt * 64 + i * 16 + lm;
        rows[i] = r < M ? r : M - 1;       // clamp tail-tile loads; stores predicated
    }
    const int n0 = nt * 32 + lm;

    f32x8 acc[4][2] = {};
    mma_pass(A1, W1, K1, rows, n0, kh, acc);
    if (A2) mma_pass(A2, W2, K2, rows, n0, kh, acc);  // fused SAGE lin_r term

    float bn[2];
    bn[0] = bias ? bias[n0] : 0.f;
    bn[1] = bias ? bias[n0 + 16] : 0.f;
#pragma unroll
    for (int i = 0; i < 4; ++i) {
#pragma unroll
        for (int r = 0; r < 8; ++r) {
            const int m = mt * 64 + i * 16 + r + kh * 8;
            if (m >= M) continue;
#pragma unroll
            for (int j = 0; j < 2; ++j) {
                float v = acc[i][j][r] + bn[j];
                if (act == 1) v = v > 0.f ? v : (expf(v) - 1.f);  // ELU
                const int n = n0 + j * 16;
                C[(size_t)m * N + n] = v;
                if (Cbf) Cbf[(size_t)m * N + n] = (bf16_t)v;
            }
        }
    }
}

// ---------------- LayerNorm + activation (wave per row, N=512) ----------------
__global__ void __launch_bounds__(256) k_ln_act(
    const float* __restrict__ in, const float* __restrict__ g, const float* __restrict__ b,
    float* __restrict__ out, bf16_t* __restrict__ outbf, int M, int act) {
    const int N = 512;
    const int lane = threadIdx.x & 31;
    const int row  = blockIdx.x * 8 + (threadIdx.x >> 5);
    if (row >= M) return;
    const float* x = in + (size_t)row * N;
    float v[16];
    float s = 0.f, s2 = 0.f;
#pragma unroll
    for (int i = 0; i < 16; ++i) {
        float t = x[lane + (i << 5)];
        v[i] = t;
        s += t;
        s2 += t * t;
    }
#pragma unroll
    for (int o = 16; o > 0; o >>= 1) {
        s  += __shfl_xor(s,  o, 32);
        s2 += __shfl_xor(s2, o, 32);
    }
    const float invn = 1.f / (float)N;
    const float mu = s * invn;
    float var = s2 * invn - mu * mu;
    var = var < 0.f ? 0.f : var;
    const float r = rsqrtf(var + 1e-5f);
#pragma unroll
    for (int i = 0; i < 16; ++i) {
        const int col = lane + (i << 5);
        float y = (v[i] - mu) * r * g[col] + b[col];
        if (act == 2)      y = 0.5f * y * (1.f + erff(y * 0.7071067811865475f)); // exact GELU
        else if (act == 3) y = y / (1.f + expf(-y));                             // SiLU
        out[(size_t)row * N + col] = y;
        if (outbf) outbf[(size_t)row * N + col] = (bf16_t)y;
    }
}

// ---------------- orchestration ----------------

static inline unsigned blks(long long n, int t) { return (unsigned)((n + t - 1) / t); }

extern "C" void kernel_launch(void* const* d_in, const int* in_sizes, int n_in,
                              void* d_out, int out_size, void* d_ws, size_t ws_size,
                              hipStream_t stream) {
    const float* x_found = (const float*)d_in[0];   // [20000,512]
    const float* x_expr  = (const float*)d_in[1];   // [20000,1024]
    const int*   edges   = (const int*)d_in[2];     // [2,640000]
    const int*   src     = edges;
    const int*   dst     = edges + NEDG;
    const float* Wl0 = (const float*)d_in[3];
    const float* bl0 = (const float*)d_in[4];
    const float* Wr0 = (const float*)d_in[5];
    const float* Wl1 = (const float*)d_in[6];
    const float* bl1 = (const float*)d_in[7];
    const float* Wr1 = (const float*)d_in[8];
    const float* Wf  = (const float*)d_in[9];
    const float* bfv = (const float*)d_in[10];
    const float* g1  = (const float*)d_in[11];
    const float* b1  = (const float*)d_in[12];
    const float* Wfu = (const float*)d_in[13];
    const float* bfu = (const float*)d_in[14];
    const float* g2  = (const float*)d_in[15];
    const float* b2  = (const float*)d_in[16];
    const float* Wl2 = (const float*)d_in[17];
    const float* bl2 = (const float*)d_in[18];
    const float* Wr2 = (const float*)d_in[19];
    float* out = (float*)d_out;                     // [20000,512]

    // workspace layout
    char* ws = (char*)d_ws;
    size_t off = 0;
    auto alloc = [&](size_t bytes) -> char* {
        char* p = ws + off;
        off += (bytes + 255) & ~(size_t)255;
        return p;
    };
    bf16_t* Wl0b = (bf16_t*)alloc((size_t)EMBD * NC * 2);
    bf16_t* Wr0b = (bf16_t*)alloc((size_t)EMBD * NC * 2);
    bf16_t* Wl1b = (bf16_t*)alloc((size_t)EMBD * EMBD * 2);
    bf16_t* Wr1b = (bf16_t*)alloc((size_t)EMBD * EMBD * 2);
    bf16_t* Wfb  = (bf16_t*)alloc((size_t)EMBD * EMBD * 2);
    bf16_t* Wfub = (bf16_t*)alloc((size_t)EMBD * EMBD * 2);
    bf16_t* Wl2b = (bf16_t*)alloc((size_t)EMBD * EMBD * 2);
    bf16_t* Wr2b = (bf16_t*)alloc((size_t)EMBD * EMBD * 2);
    float*  DEG  = (float*)alloc((size_t)NG * 4);
    bf16_t* XEb  = (bf16_t*)alloc((size_t)NG * NC * 2);
    bf16_t* XFb  = (bf16_t*)alloc((size_t)NG * EMBD * 2);
    float*  AGG  = (float*)alloc((size_t)NG * NC * 4);     // reused: agg0/agg1/f_pre/agg2
    bf16_t* MEANb= (bf16_t*)alloc((size_t)NG * NC * 2);    // reused: mean0/mean1/fused/mean2
    float*  H0   = (float*)alloc((size_t)NG * EMBD * 4);   // reused: h0 / fu_pre / fo
    bf16_t* H0b  = (bf16_t*)alloc((size_t)NG * EMBD * 2);  // reused: h0_bf / fo_bf
    float*  H1   = (float*)alloc((size_t)NG * EMBD * 4);   // h1 (kept until fusion)

    const int T = 256;
    const long long nXE = (long long)NG * NC;    // 20.48M
    const long long nXF = (long long)NG * EMBD;  // 10.24M
    // waves = ceil(M/64) * (N/32); 8 waves per block
    const unsigned gemm_blocks = (unsigned)((((NG + 63) / 64) * (EMBD / 32) + 7) / 8);
    const unsigned ln_blocks = (NG + 7) / 8;
    const int FS_NC = 10, FS_EMB = 9;            // log2 feature widths

    // --- convert operands to bf16 ---
    k_cvt<<<blks(nXE, T), T, 0, stream>>>(x_expr, XEb, (int)nXE);
    k_cvt<<<blks(nXF, T), T, 0, stream>>>(x_found, XFb, (int)nXF);
    k_cvt<<<blks(EMBD * NC, T), T, 0, stream>>>(Wl0, Wl0b, EMBD * NC);
    k_cvt<<<blks(EMBD * NC, T), T, 0, stream>>>(Wr0, Wr0b, EMBD * NC);
    k_cvt<<<blks(EMBD * EMBD, T), T, 0, stream>>>(Wl1, Wl1b, EMBD * EMBD);
    k_cvt<<<blks(EMBD * EMBD, T), T, 0, stream>>>(Wr1, Wr1b, EMBD * EMBD);
    k_cvt<<<blks(EMBD * EMBD, T), T, 0, stream>>>(Wf, Wfb, EMBD * EMBD);
    k_cvt<<<blks(EMBD * EMBD, T), T, 0, stream>>>(Wfu, Wfub, EMBD * EMBD);
    k_cvt<<<blks(EMBD * EMBD, T), T, 0, stream>>>(Wl2, Wl2b, EMBD * EMBD);
    k_cvt<<<blks(EMBD * EMBD, T), T, 0, stream>>>(Wr2, Wr2b, EMBD * EMBD);

    // --- degrees ---
    k_zero<<<blks(NG, T), T, 0, stream>>>(DEG, NG);
    k_deg<<<blks(NEDG, T), T, 0, stream>>>(dst, DEG, NEDG);

    // --- SAGE layer 0 (1024 -> 512), ELU ---
    k_zero<<<blks(nXE, T), T, 0, stream>>>(AGG, (int)nXE);
    k_scatter4<<<blks((long long)NEDG * (NC / 4), T), T, 0, stream>>>(x_expr, src, dst, AGG, NEDG, FS_NC);
    k_mean_cvt<<<blks(nXE, T), T, 0, stream>>>(AGG, DEG, MEANb, NG, FS_NC);
    k_wmma_gemm<<<gemm_blocks, T, 0, stream>>>(MEANb, Wl0b, NC, XEb, Wr0b, NC,
                                               bl0, H0, H0b, NG, EMBD, /*ELU*/1);

    // --- SAGE layer 1 (512 -> 512), ELU ---
    k_zero<<<blks(nXF, T), T, 0, stream>>>(AGG, (int)nXF);
    k_scatter4<<<blks((long long)NEDG * (EMBD / 4), T), T, 0, stream>>>(H0, src, dst, AGG, NEDG, FS_EMB);
    k_mean_cvt<<<blks(nXF, T), T, 0, stream>>>(AGG, DEG, MEANb, NG, FS_EMB);
    k_wmma_gemm<<<gemm_blocks, T, 0, stream>>>(MEANb, Wl1b, EMBD, H0b, Wr1b, EMBD,
                                               bl1, H1, (bf16_t*)nullptr, NG, EMBD, /*ELU*/1);

    // --- foundation branch: Linear -> LN -> exact GELU (f_pre in AGG, LN in place) ---
    k_wmma_gemm<<<gemm_blocks, T, 0, stream>>>(XFb, Wfb, EMBD, (bf16_t*)nullptr, (bf16_t*)nullptr, 0,
                                               bfv, AGG, (bf16_t*)nullptr, NG, EMBD, /*none*/0);
    k_ln_act<<<ln_blocks, T, 0, stream>>>(AGG, g1, b1, AGG, (bf16_t*)nullptr, NG, /*GELU*/2);

    // --- fuse: fused = h1 + f ; Linear -> LN -> SiLU ---
    k_add_cvt<<<blks(nXF, T), T, 0, stream>>>(H1, AGG, MEANb, (int)nXF);
    k_wmma_gemm<<<gemm_blocks, T, 0, stream>>>(MEANb, Wfub, EMBD, (bf16_t*)nullptr, (bf16_t*)nullptr, 0,
                                               bfu, H0, (bf16_t*)nullptr, NG, EMBD, /*none*/0);
    k_ln_act<<<ln_blocks, T, 0, stream>>>(H0, g2, b2, H0, H0b, NG, /*SiLU*/3);

    // --- final SAGE layer (512 -> 512), ELU -> d_out ---
    k_zero<<<blks(nXF, T), T, 0, stream>>>(AGG, (int)nXF);
    k_scatter4<<<blks((long long)NEDG * (EMBD / 4), T), T, 0, stream>>>(H0, src, dst, AGG, NEDG, FS_EMB);
    k_mean_cvt<<<blks(nXF, T), T, 0, stream>>>(AGG, DEG, MEANb, NG, FS_EMB);
    k_wmma_gemm<<<gemm_blocks, T, 0, stream>>>(MEANb, Wl2b, EMBD, H0b, Wr2b, EMBD,
                                               bl2, out, (bf16_t*)nullptr, NG, EMBD, /*ELU*/1);
}
